// PointNetSetAbstraction_66838281060813
// MI455X (gfx1250) — matrix operand order repaired and checked by reading, not compiled, not run on hardware
//
#include <hip/hip_runtime.h>
#include <hip/hip_bf16.h>
#include <cstdint>

// ---------------------------------------------------------------------------
// PointNet++ SetAbstraction for MI455X (gfx1250, wave32, WMMA + TDM).
//   B=16, N=4096, S=1024, K=32, radius=0.2; channels 67 -> 64 -> 64 -> 128.
// GEMMs: v_wmma_f32_16x16x32_f16 (f32 accumulate), f16 activations (halves
// the ~1.2 GB of activation traffic; op is memory/latency bound vs 23.3 TB/s
// HBM + 192 MB L2). Weights staged block-wide into LDS with one TDM DMA.
// ---------------------------------------------------------------------------

typedef __attribute__((ext_vector_type(16))) _Float16 v16h;
typedef __attribute__((ext_vector_type(8)))  _Float16 v8h;
typedef __attribute__((ext_vector_type(8)))  float    v8f;

typedef __attribute__((ext_vector_type(4))) unsigned int u32x4;
typedef __attribute__((ext_vector_type(8))) int          i32x8;
typedef __attribute__((ext_vector_type(4))) int          i32x4;

#define B_   16
#define N_   4096
#define S_   1024
#define K_   32
#define M_   (B_ * S_ * K_)     // 524288 feature rows
#define R2_  0.04f              // radius^2
#define NRED 256                // stats reduction blocks (deterministic 2-stage)

// ---------------------------------------------------------------------------
// TDM: one scalar DMA of a contiguous f16 block (1-D tensor descriptor) into
// LDS. Arity differs by toolchain (ROCm7.2 clang-22: 5 args; amdgpu-toolchain
// clang-23: 6 args) -- the TDM header only ships with the 6-arg toolchain.
// ---------------------------------------------------------------------------
#if __has_include(<hip/amd_detail/amd_gfx1250_TDM.h>)
#define TDM_SIX_ARGS 1
#else
#define TDM_SIX_ARGS 0
#endif

template <int NELEM>
__device__ __forceinline__ void tdm_copy_to_lds(const _Float16* __restrict__ src,
                                                _Float16* ldsDst)
{
    static_assert(NELEM < 65536, "tile_dim0 is 16-bit");
    const unsigned long long ga = (unsigned long long)(const void*)src;
    const unsigned ldsOff = (unsigned)(uintptr_t)ldsDst;   // low 32 bits = LDS offset

    u32x4 g0;
    g0[0] = 1u;                                            // count=1, user descriptor
    g0[1] = ldsOff;                                        // lds_addr (bytes)
    g0[2] = (unsigned)(ga & 0xFFFFFFFFull);                // global_addr[31:0]
    g0[3] = (unsigned)((ga >> 32) & 0x1FFFFFFull)          // global_addr[56:32]
          | (2u << 30);                                    // type = 2 ("image")

    i32x8 g1 = {0, 0, 0, 0, 0, 0, 0, 0};
    const unsigned td0 = (unsigned)NELEM;                  // 1-D, f16 elements
    g1[0] = (int)(1u << 16);                               // data_size = 1 (2 bytes)
    g1[1] = (int)((td0 & 0xFFFFu) << 16);                  // tensor_dim0[15:0]
    g1[2] = (int)((td0 >> 16) | (1u << 16));               // tensor_dim0[31:16], dim1=1
    g1[3] = (int)(td0 << 16);                              // tile_dim0 = NELEM

    const i32x4 z4 = {0, 0, 0, 0};
#if TDM_SIX_ARGS
    const i32x8 z8 = {0, 0, 0, 0, 0, 0, 0, 0};
    __builtin_amdgcn_tensor_load_to_lds(g0, g1, z4, z4, z8, 0);
#else
    __builtin_amdgcn_tensor_load_to_lds(g0, g1, z4, z4, 0);
#endif
}

// ---------------------------------------------------------------------------
// 1) Farthest point sampling. One block/batch; cloud in LDS, running min-
//    distance in registers. Argmax per step: wave32 shfl_xor reduction (no
//    barriers) + 8-entry cross-wave fold -> 2 barriers/step (was 9).
//    Tie-break matches jnp.argmax (lowest index wins).
// ---------------------------------------------------------------------------
__global__ void fps_kernel(const float* __restrict__ xyz,
                           float* __restrict__ newXyz)
{
    const int b    = blockIdx.x;
    const int t    = threadIdx.x;           // 0..255
    const int lane = t & 31;
    const int wid  = t >> 5;                // 0..7

    __shared__ float px[N_], py[N_], pz[N_];
    __shared__ float rv[8];
    __shared__ int   ri[8];
    __shared__ int   sFar;

    const float* base = xyz + (size_t)b * N_ * 3;
    for (int i = t; i < N_; i += 256) {
        px[i] = base[i * 3 + 0];
        py[i] = base[i * 3 + 1];
        pz[i] = base[i * 3 + 2];
    }
    float dist[16];
#pragma unroll
    for (int j = 0; j < 16; ++j) dist[j] = __builtin_inff();
    __syncthreads();

    int far = 0;
    for (int it = 0; it < S_; ++it) {
        const float cx = px[far], cy = py[far], cz = pz[far];
        if (t == 0) {
            float* o = newXyz + ((size_t)b * S_ + it) * 3;
            o[0] = cx; o[1] = cy; o[2] = cz;
        }
        float best = -1.0f; int bestI = 0;
#pragma unroll
        for (int j = 0; j < 16; ++j) {
            const int p = t + j * 256;
            const float dx = px[p] - cx, dy = py[p] - cy, dz = pz[p] - cz;
            const float d = dx * dx + dy * dy + dz * dz;
            dist[j] = fminf(dist[j], d);
            if (dist[j] > best) { best = dist[j]; bestI = p; }  // strict >: lowest idx
        }
#pragma unroll
        for (int off = 16; off > 0; off >>= 1) {
            const float ov = __shfl_xor(best, off, 32);
            const int   oi = __shfl_xor(bestI, off, 32);
            if (ov > best || (ov == best && oi < bestI)) { best = ov; bestI = oi; }
        }
        if (lane == 0) { rv[wid] = best; ri[wid] = bestI; }
        __syncthreads();
        if (t == 0) {
            float bv = rv[0]; int bi = ri[0];
#pragma unroll
            for (int w = 1; w < 8; ++w)
                if (rv[w] > bv || (rv[w] == bv && ri[w] < bi)) { bv = rv[w]; bi = ri[w]; }
            sFar = bi;
        }
        __syncthreads();
        far = sFar;
    }
}

// ---------------------------------------------------------------------------
// 2) Ball query (first K in-radius indices, index order, padded with first
//    hit) + gather + concat + f16 pack into X0 (channel pad 67 -> 96).
//    One wave per (b,s): ballot + prefix-popcount ordered append.
// ---------------------------------------------------------------------------
__global__ void ballquery_gather(const float* __restrict__ xyz,
                                 const float* __restrict__ points,
                                 const float* __restrict__ newXyz,
                                 _Float16* __restrict__ X0)
{
    const int bs   = blockIdx.x;            // b*S + s
    const int b    = bs >> 10;
    const int lane = threadIdx.x;           // 0..31 (wave32)

    __shared__ int list[K_];

    const float* q  = newXyz + (size_t)bs * 3;
    const float qx = q[0], qy = q[1], qz = q[2];
    const float* base = xyz + (size_t)b * N_ * 3;

    int have = 0;
    for (int c0 = 0; c0 < N_ && have < K_; c0 += 32) {
        const int i = c0 + lane;
        const float dx = base[i * 3 + 0] - qx;
        const float dy = base[i * 3 + 1] - qy;
        const float dz = base[i * 3 + 2] - qz;
        const bool in = (dx * dx + dy * dy + dz * dz) <= R2_;
        const unsigned m = (unsigned)__ballot((int)in);   // wave32: low 32 bits
        const int rank = __popc(m & ((1u << lane) - 1u));
        if (in && (have + rank) < K_) list[have + rank] = i;
        have += __popc(m);
        if (have > K_) have = K_;
    }
    __syncthreads();
    const int first = (have > 0) ? list[0] : 0;
    for (int k = have + lane; k < K_; k += 32) list[k] = first;
    __syncthreads();

    _Float16* out = X0 + (size_t)bs * K_ * 96;
    for (int k = 0; k < K_; ++k) {
        const int id = list[k];
        _Float16* row = out + k * 96;
        for (int cc = lane; cc < 96; cc += 32) {
            float v;
            if (cc < 3)       v = base[id * 3 + cc] - q[cc];
            else if (cc < 67) v = points[((size_t)b * 64 + (cc - 3)) * N_ + id];
            else              v = 0.0f;
            row[cc] = (_Float16)v;
        }
    }
}

// 3) Weight convert: (O, C) f32 -> (O, CPAD) f16 row-major, zero pad.
__global__ void convert_w(const float* __restrict__ w, _Float16* __restrict__ wf,
                          int rows, int cols, int cpad)
{
    const int i = blockIdx.x * blockDim.x + threadIdx.x;
    if (i >= rows * cpad) return;
    const int r = i / cpad, c = i % cpad;
    wf[i] = (c < cols) ? (_Float16)w[r * cols + c] : (_Float16)0.0f;
}

// ---------------------------------------------------------------------------
// 4) GEMM  Y[M x COUT] = X[M x CPAD] * Wf^T,  Wf is (COUT x CPAD) row-major.
//    4 waves/block, one 16x16 tile each. Weights staged into LDS by a single
//    TDM DMA (wave 0) -> s_wait_tensorcnt -> barrier; B-fragments then come
//    from LDS banks (broadcast to all 4 waves), A-fragments via b128 global.
// ---------------------------------------------------------------------------
template <int CPAD, int COUT>
__global__ void gemm_wmma(const _Float16* __restrict__ X,
                          const _Float16* __restrict__ Wf,
                          _Float16* __restrict__ Y)
{
    __shared__ __align__(64) _Float16 ldsW[COUT * CPAD];

    const int wave = threadIdx.x >> 5;          // 0..3
    const int lane = threadIdx.x & 31;
    const int l16  = lane & 15;
    const int hi   = lane >> 4;                 // lane-half select
    const int rowBase = blockIdx.x * 16;
    const int colBase = blockIdx.y * 64 + wave * 16;

    if (threadIdx.x < 32) {
        tdm_copy_to_lds<COUT * CPAD>(Wf, ldsW); // async DMA, TENSORcnt++
        __builtin_amdgcn_s_wait_tensorcnt(0);
    }
    __syncthreads();

    constexpr int KSTEPS = CPAD / 32;

    // B fragments from LDS: lane n holds 16 contiguous K halves (k0 + {0|16}).
    v16h bfrag[KSTEPS];
    const _Float16* wrow = ldsW + (size_t)(colBase + l16) * CPAD + hi * 16;
#pragma unroll
    for (int s = 0; s < KSTEPS; ++s)
        bfrag[s] = *(const v16h*)(wrow + s * 32);

    const _Float16* xrow = X + (size_t)(rowBase + l16) * CPAD + hi * 16;
    __builtin_prefetch(xrow, 0, 3);             // global_prefetch_b8

    v8f acc = {};
#pragma unroll
    for (int s = 0; s < KSTEPS; ++s) {
        const v8h lo  = *(const v8h*)(xrow + s * 32);
        const v8h hi8 = *(const v8h*)(xrow + s * 32 + 8);
        const v16h a = __builtin_shufflevector(lo, hi8,
            0, 1, 2, 3, 4, 5, 6, 7, 8, 9, 10, 11, 12, 13, 14, 15);
        acc = __builtin_amdgcn_wmma_f32_16x16x32_f16(
            false, a, false, bfrag[s], (short)0, acc, false, false);
    }

    _Float16* yrow = Y + (size_t)(rowBase + hi * 8) * COUT + colBase + l16;
#pragma unroll
    for (int r = 0; r < 8; ++r)
        yrow[(size_t)r * COUT] = (_Float16)acc[r];
}

// ---------------------------------------------------------------------------
// 5) BN stats: deterministic two-stage reduction (no float atomics).
// ---------------------------------------------------------------------------
__global__ void reduce_stats(const _Float16* __restrict__ Y, int C,
                             float* __restrict__ partial /* [2*NRED*C] */)
{
    const int t   = threadIdx.x;            // 0..255
    const int col = t % C;
    const int grp = t / C;
    const int groups = 256 / C;             // 4 for C=64, 2 for C=128

    float s = 0.0f, q = 0.0f;
    for (int r = blockIdx.x * groups + grp; r < M_; r += NRED * groups) {
        const float v = (float)Y[(size_t)r * C + col];
        s += v; q += v * v;
    }
    __shared__ float ss[256], sq[256];
    ss[t] = s; sq[t] = q;
    __syncthreads();
    for (int g = groups >> 1; g > 0; g >>= 1) {
        if (grp < g) { ss[t] += ss[t + g * C]; sq[t] += sq[t + g * C]; }
        __syncthreads();
    }
    if (grp == 0) {
        partial[(size_t)blockIdx.x * C + col]            = ss[t];
        partial[(size_t)NRED * C + blockIdx.x * C + col] = sq[t];
    }
}

__global__ void bn_finalize(const float* __restrict__ partial,
                            const float* __restrict__ g,
                            const float* __restrict__ bb,
                            int C, float* __restrict__ scaleShift /* [2C] */)
{
    const int c = threadIdx.x;
    if (c >= C) return;
    float s = 0.0f, q = 0.0f;
    for (int k = 0; k < NRED; ++k) {
        s += partial[(size_t)k * C + c];
        q += partial[(size_t)NRED * C + k * C + c];
    }
    const float mean = s / (float)M_;
    const float var  = q / (float)M_ - mean * mean;
    const float sc   = g[c] * rsqrtf(var + 1e-5f);
    scaleShift[c]     = sc;
    scaleShift[C + c] = bb[c] - mean * sc;
}

// 6) Normalize + ReLU in place, 8 f16 per thread (b128 load/store).
__global__ void bn_relu_apply(_Float16* __restrict__ Y,
                              const float* __restrict__ scaleShift,
                              int C, size_t total8)
{
    const size_t i = (size_t)blockIdx.x * blockDim.x + threadIdx.x;
    if (i >= total8) return;
    const size_t e = i * 8;
    const int c0 = (int)(e % (size_t)C);    // C multiple of 8 -> c0..c0+7 in range
    v8h v = *(v8h*)(Y + e);
#pragma unroll
    for (int j = 0; j < 8; ++j) {
        const float f = (float)v[j] * scaleShift[c0 + j] + scaleShift[C + c0 + j];
        v[j] = (_Float16)fmaxf(f, 0.0f);
    }
    *(v8h*)(Y + e) = v;
}

// ---------------------------------------------------------------------------
// 7) Fused BN+ReLU+max over K + transpose to (B, 128, S), f32 out. Saves a
//    full 268 MB normalize pass on the last layer.
// ---------------------------------------------------------------------------
__global__ void maxpool_transpose(const _Float16* __restrict__ Y2,
                                  const float* __restrict__ scaleShift,
                                  float* __restrict__ out)
{
    const int bs = blockIdx.x;              // b*S + s
    const int c  = threadIdx.x;             // 0..127
    const float sc = scaleShift[c], sh = scaleShift[128 + c];
    const _Float16* p = Y2 + (size_t)bs * K_ * 128 + c;
    float m = -__builtin_inff();
    for (int k = 0; k < K_; ++k) m = fmaxf(m, (float)p[(size_t)k * 128] * sc + sh);
    m = fmaxf(m, 0.0f);                     // ReLU commutes with max
    const int b = bs >> 10, s = bs & 1023;
    out[(size_t)(b * 128 + c) * S_ + s] = m;
}

// ---------------------------------------------------------------------------
// Host-side orchestration.
// ---------------------------------------------------------------------------
extern "C" void kernel_launch(void* const* d_in, const int* in_sizes, int n_in,
                              void* d_out, int out_size, void* d_ws, size_t ws_size,
                              hipStream_t stream)
{
    (void)in_sizes; (void)n_in; (void)out_size; (void)ws_size;

    const float* xyz    = (const float*)d_in[0];
    const float* points = (const float*)d_in[1];
    const float* w0 = (const float*)d_in[2];
    const float* g0 = (const float*)d_in[3];
    const float* b0 = (const float*)d_in[4];
    const float* w1 = (const float*)d_in[5];
    const float* g1 = (const float*)d_in[6];
    const float* b1 = (const float*)d_in[7];
    const float* w2 = (const float*)d_in[8];
    const float* g2 = (const float*)d_in[9];
    const float* b2 = (const float*)d_in[10];

    float* out = (float*)d_out;
    float* newXyz  = out;                        // (B, S, 3)
    float* featOut = out + (size_t)B_ * S_ * 3;  // (B, 128, S)

    // Workspace layout (256B-aligned offsets), ~235 MB total:
    char* ws = (char*)d_ws;
    _Float16* Wf0 = (_Float16*)(ws + 0);                 //  64 x 96  f16
    _Float16* Wf1 = (_Float16*)(ws + 12288);             //  64 x 64  f16
    _Float16* Wf2 = (_Float16*)(ws + 20480);             // 128 x 64  f16
    float*    partial = (float*)(ws + 36864);            // 2*NRED*128 f32
    float*    ss0 = (float*)(ws + 299008);               // 2*64
    float*    ss1 = (float*)(ws + 300032);
    float*    ss2 = (float*)(ws + 301056);               // 2*128
    _Float16* bufA = (_Float16*)(ws + 302080);                        // M*96 f16
    _Float16* bufB = (_Float16*)(ws + 302080 + (size_t)M_ * 96 * 2);  // M*128 f16

    _Float16* X0 = bufA;            // M x 96
    _Float16* Y0 = bufB;            // M x 64
    _Float16* Y1 = bufA;            // M x 64  (X0 dead after gemm0)
    _Float16* Y2 = bufB;            // M x 128 (Y0 dead after gemm1)

    fps_kernel<<<B_, 256, 0, stream>>>(xyz, newXyz);

    convert_w<<<(64 * 96 + 255) / 256, 256, 0, stream>>>(w0, Wf0, 64, 67, 96);
    convert_w<<<(64 * 64 + 255) / 256, 256, 0, stream>>>(w1, Wf1, 64, 64, 64);
    convert_w<<<(128 * 64 + 255) / 256, 256, 0, stream>>>(w2, Wf2, 128, 64, 64);

    ballquery_gather<<<B_ * S_, 32, 0, stream>>>(xyz, points, newXyz, X0);

    // Layer 0: 96(->67 real) -> 64
    gemm_wmma<96, 64><<<dim3(M_ / 16, 1), 128, 0, stream>>>(X0, Wf0, Y0);
    reduce_stats<<<NRED, 256, 0, stream>>>(Y0, 64, partial);
    bn_finalize<<<1, 128, 0, stream>>>(partial, g0, b0, 64, ss0);
    bn_relu_apply<<<(unsigned)(((size_t)M_ * 64 / 8 + 255) / 256), 256, 0, stream>>>(
        Y0, ss0, 64, (size_t)M_ * 64 / 8);

    // Layer 1: 64 -> 64
    gemm_wmma<64, 64><<<dim3(M_ / 16, 1), 128, 0, stream>>>(Y0, Wf1, Y1);
    reduce_stats<<<NRED, 256, 0, stream>>>(Y1, 64, partial);
    bn_finalize<<<1, 128, 0, stream>>>(partial, g1, b1, 64, ss1);
    bn_relu_apply<<<(unsigned)(((size_t)M_ * 64 / 8 + 255) / 256), 256, 0, stream>>>(
        Y1, ss1, 64, (size_t)M_ * 64 / 8);

    // Layer 2: 64 -> 128 (BN+ReLU fused into maxpool read)
    gemm_wmma<64, 128><<<dim3(M_ / 16, 2), 128, 0, stream>>>(Y1, Wf2, Y2);
    reduce_stats<<<NRED, 256, 0, stream>>>(Y2, 128, partial);
    bn_finalize<<<1, 128, 0, stream>>>(partial, g2, b2, 128, ss2);

    maxpool_transpose<<<B_ * S_, 128, 0, stream>>>(Y2, ss2, featOut);
}
